// RotationAwareLoss_66262755443031
// MI455X (gfx1250) — compile-verified, matrix operand order checked by reading, not run on hardware
//
#include <hip/hip_runtime.h>
#include <hip/hip_bf16.h>
#include <math.h>

// ---------------------------------------------------------------------------
// RotationAwareLoss on MI455X (gfx1250, wave32).
// Memory-bound streaming loss (~35 MB traffic); rotated-IoU only on the 19x19
// level (M=1083<2000). Final reduction uses V_WMMA_F32_16X16X4_F32 with a
// ones-B matrix (exact f32 row-sum reduction on one wave).
// ---------------------------------------------------------------------------

#define PI_F 3.14159265358979323846f

__constant__ float c_anchors[18] = {
    18.7807f, 33.4659f,  28.8912f, 61.7536f,  48.6849f, 68.3897f,
    45.0668f, 101.4673f, 63.0952f, 113.5382f, 81.3909f, 134.4554f,
    91.7364f, 144.9949f, 137.5189f, 178.4791f, 194.4429f, 250.7985f};

// ---- workspace layout (bytes) ----
#define WS_CNT       0        // 64 ints
#define WS_VALIDANY  256      // 64 ints
#define WS_PENMASK   512      // 64*1083 bytes (level-L penalty mask)
#define WS_PARTIALS  70144    // 8192 floats (zero-padded partial sums)
#define NPART        8192
#define OFF_L        0        // 320 blocks (5 per batch x 64)
#define OFF_M        320      // 1083 blocks
#define OFF_S        1403     // 4333 blocks
#define OFF_POS      5736     // 1 slot

// ---------------- helpers ----------------
__device__ __forceinline__ float sigmoidf_(float x) { return 1.f / (1.f + __expf(-x)); }
// bce(x,t) = max(x,0) - x*t + softplus(-|x|)
__device__ __forceinline__ float bce_(float x, float t) {
  return fmaxf(x, 0.f) - x * t + log1pf(__expf(-fabsf(x)));
}
__device__ __forceinline__ float softplus_(float x) { return bce_(x, 0.f); }
__device__ __forceinline__ float clamp01_(float x) { return fminf(fmaxf(x, 0.f), 1.f); }

__device__ __forceinline__ float wave_reduce(float v) {
  #pragma unroll
  for (int off = 16; off > 0; off >>= 1) v += __shfl_xor(v, off, 32);
  return v;
}
// valid result on thread 0; blockDim.x must be a multiple of 32, <= 256
__device__ __forceinline__ float block_reduce(float v) {
  __shared__ float s[8];
  int lane = threadIdx.x & 31, w = threadIdx.x >> 5;
  v = wave_reduce(v);
  if (lane == 0) s[w] = v;
  __syncthreads();
  int nw = (blockDim.x + 31) >> 5;
  v = (threadIdx.x < (unsigned)nw) ? s[threadIdx.x] : 0.f;
  if (w == 0) v = wave_reduce(v);
  return v;
}

__device__ int best_anchor(float tw, float th) {
  float best = -1.f; int bi = 0;
  #pragma unroll
  for (int k = 0; k < 9; k++) {
    float aw = c_anchors[2 * k] * (1.f / 608.f);
    float ah = c_anchors[2 * k + 1] * (1.f / 608.f);
    float inter = fminf(tw, aw) * fminf(th, ah);
    float uni = tw * th + aw * ah - inter;
    float iou = inter / (uni + 1e-16f);
    if (iou > best) { best = iou; bi = k; }
  }
  return bi;
}

struct Box { float cx, cy, w, h, ang; };

__device__ void rect_corners(const Box& b, float px[4], float py[4]) {
  float th = b.ang * (PI_F / 180.f);
  float ct = __cosf(th), st = __sinf(th);
  const float dxs[4] = {0.5f, 0.5f, -0.5f, -0.5f};
  const float dys[4] = {-0.5f, 0.5f, 0.5f, -0.5f};
  #pragma unroll
  for (int i = 0; i < 4; i++) {
    float X = dxs[i] * b.w, Y = dys[i] * b.h;
    px[i] = b.cx + X * ct - Y * st;
    py[i] = b.cy + X * st + Y * ct;
  }
}

// Sutherland-Hodgman: clip quad A by convex quad B, return intersection area.
__device__ float clip_area(const float pax[4], const float pay[4],
                           const float pbx[4], const float pby[4]) {
  float px[8], py[8];
  #pragma unroll
  for (int i = 0; i < 4; i++) { px[i] = pax[i]; py[i] = pay[i]; }
  #pragma unroll
  for (int i = 4; i < 8; i++) { px[i] = 0.f; py[i] = 0.f; }
  int n = 4;
  for (int e = 0; e < 4; e++) {
    float p1x = pbx[e], p1y = pby[e];
    int e2 = (e + 1) & 3;
    float ex = pbx[e2] - p1x, ey = pby[e2] - p1y;
    float ox[8], oy[8];
    #pragma unroll
    for (int i = 0; i < 8; i++) { ox[i] = 0.f; oy[i] = 0.f; }
    int m = 0;
    for (int i = 0; i < 8; i++) {
      if (i >= n) continue;
      float cx = px[i], cy = py[i];
      int j = (i + 1 >= n) ? 0 : (i + 1);
      float nx = px[j], ny = py[j];
      float sc = ex * (cy - p1y) - ey * (cx - p1x);
      float sn = ex * (ny - p1y) - ey * (nx - p1x);
      bool kc = (sc >= 0.f);
      if (kc && m < 8) { ox[m] = cx; oy[m] = cy; m++; }
      if ((kc != (sn >= 0.f)) && m < 8) {
        float denom = sc - sn;
        float t = sc / ((denom == 0.f) ? 1.f : denom);
        ox[m] = cx + t * (nx - cx);
        oy[m] = cy + t * (ny - cy);
        m++;
      }
    }
    #pragma unroll
    for (int i = 0; i < 8; i++) { px[i] = ox[i]; py[i] = oy[i]; }
    n = m;
  }
  float s = 0.f;
  for (int i = 0; i < 8; i++) {
    if (i >= n) continue;
    int j = (i + 1 >= n) ? 0 : (i + 1);
    s += px[i] * py[j] - px[j] * py[i];
  }
  return 0.5f * fabsf(s);
}

__device__ float rot_iou(const Box& a, const Box& b) {
  float pax[4], pay[4], pbx[4], pby[4];
  rect_corners(a, pax, pay);
  rect_corners(b, pbx, pby);
  float inter = clip_area(pax, pay, pbx, pby);
  float uni = a.w * a.h + b.w * b.h - inter;
  return inter / (uni + 1e-16f);
}

// ---------------- kernels ----------------
__global__ void k_init(int* cnt, int* validany, float* partials) {
  int id = blockIdx.x * blockDim.x + threadIdx.x;
  if (id < 64) { cnt[id] = 0; validany[id] = 0; }
  if (id < NPART) partials[id] = 0.f;
}

// per-batch: any valid label whose best anchor belongs to level L (anchors 6..8)
__global__ void k_prep(const float* __restrict__ labels, int* validany) {
  int b = blockIdx.x * blockDim.x + threadIdx.x;
  if (b >= 64) return;
  int any = 0;
  for (int l = 0; l < 20; l++) {
    const float* lab = labels + (b * 20 + l) * 5;
    if (lab[0] + lab[1] + lab[2] + lab[3] > 0.f) {
      if (best_anchor(lab[2], lab[3]) >= 6) any = 1;
    }
  }
  validany[b] = any;
}

// count level-L cells with sigmoid(conf) > 0.001 per batch
__global__ void k_countL(const float* __restrict__ det, int* cnt) {
  int id = blockIdx.x * blockDim.x + threadIdx.x;
  if (id >= 64 * 1083) return;
  int b = id / 1083, r = id % 1083;
  int a = r / 361, yx = r % 361;
  float p5 = det[(b * 18 + a * 6 + 5) * 361 + yx];
  if (sigmoidf_(p5) > 0.001f) atomicAdd(&cnt[b], 1);
}

// level L: fused rotated-IoU ignore mask + negative conf BCE (softplus)
__global__ void k_negL(const float* __restrict__ det, const float* __restrict__ labels,
                       const int* __restrict__ cnt, const int* __restrict__ validany,
                       unsigned char* penMask, float* partials) {
  __shared__ float s_lab[100];
  int b = blockIdx.y;
  __builtin_prefetch(labels + b * 100, 0, 0);  // global_prefetch_b8
  if (threadIdx.x < 100) s_lab[threadIdx.x] = labels[b * 100 + threadIdx.x];
  __syncthreads();

  int cell = blockIdx.x * blockDim.x + threadIdx.x;
  float partial = 0.f;
  if (cell < 1083) {
    int a = cell / 361, rem = cell % 361;
    int y = rem / 19, x = rem % 19;
    int base = (b * 18 + a * 6) * 361 + y * 19 + x;
    float p0 = det[base], p1 = det[base + 361];
    float p2 = det[base + 2 * 361], p3 = det[base + 3 * 361];
    float p4 = det[base + 4 * 361], p5 = det[base + 5 * 361];

    float conf = sigmoidf_(p5);
    int c = cnt[b];
    bool apply = (c > 0) && (c < 2000) && (validany[b] != 0);
    bool ignore = false;
    if (apply && conf > 0.001f) {
      Box pb;
      pb.cx = clamp01_((sigmoidf_(p0) + (float)x) * (1.f / 19.f));
      pb.cy = clamp01_((sigmoidf_(p1) + (float)y) * (1.f / 19.f));
      pb.w  = clamp01_(__expf(p2) * c_anchors[2 * (6 + a)] * (1.f / 608.f));
      pb.h  = clamp01_(__expf(p3) * c_anchors[2 * (6 + a) + 1] * (1.f / 608.f));
      pb.ang = sigmoidf_(p4) * 360.f - 180.f;
      float best = 0.f;
      for (int l = 0; l < 20; l++) {
        float lx = s_lab[l * 5], ly = s_lab[l * 5 + 1];
        float lw = s_lab[l * 5 + 2], lh = s_lab[l * 5 + 3], la = s_lab[l * 5 + 4];
        if (lx + ly + lw + lh > 0.f) {
          Box gt; gt.cx = lx; gt.cy = ly; gt.w = lw; gt.h = lh; gt.ang = la;
          best = fmaxf(best, rot_iou(pb, gt));
        }
      }
      ignore = best > 0.6f;
    }
    penMask[b * 1083 + cell] = ignore ? 0 : 1;
    if (!ignore) partial = softplus_(p5);
  }
  float bs = block_reduce(partial);
  if (threadIdx.x == 0) partials[OFF_L + b * gridDim.x + blockIdx.x] = bs;
}

// levels M/S: penalty==1 everywhere -> sum softplus(conf_logit)
__global__ void k_neg_conf(const float* __restrict__ det, float* partials, int off, int ns) {
  int s2 = ns * ns, cpb = 3 * s2;
  int id = blockIdx.x * blockDim.x + threadIdx.x;
  float partial = 0.f;
  if (id < 64 * cpb) {
    int b = id / cpb, r = id % cpb;
    int a = r / s2, yx = r % s2;
    partial = softplus_(det[(b * 18 + a * 6 + 5) * s2 + yx]);
  }
  float bs = block_reduce(partial);
  if (threadIdx.x == 0) partials[off + blockIdx.x] = bs;
}

// positives: one thread per batch; serial over labels with last-write-wins dedupe
// (emulates JAX scatter with duplicate indices), then per-cell loss + conf fixup.
__global__ void k_pos(const float* __restrict__ detL, const float* __restrict__ detM,
                      const float* __restrict__ detS, const float* __restrict__ labels,
                      const unsigned char* __restrict__ penL, float* partials) {
  int b = threadIdx.x;
  float total = 0.f;
  if (b < 64) {
    int keys[20], lv_[20], an_[20], tj_[20], ti_[20];
    float tv[20][5];
    int nent = 0;
    for (int l = 0; l < 20; l++) {
      const float* lab = labels + (b * 20 + l) * 5;
      float lx = lab[0], ly = lab[1], lw = lab[2], lh = lab[3], la = lab[4];
      if (!(lx + ly + lw + lh > 0.f)) continue;
      int ba = best_anchor(lw, lh);
      int lvl = (ba >= 6) ? 0 : ((ba >= 3) ? 1 : 2);  // 0=L(19) 1=M(38) 2=S(76)
      int ns = (lvl == 0) ? 19 : ((lvl == 1) ? 38 : 76);
      float tx = lx * (float)ns, ty = ly * (float)ns;
      int ti = (int)tx, tj = (int)ty;
      int bn = ba % 3;
      float aw = c_anchors[2 * ba] * (1.f / 608.f);
      float ah = c_anchors[2 * ba + 1] * (1.f / 608.f);
      float t0 = tx - floorf(tx), t1 = ty - floorf(ty);
      float t2 = logf(lw / aw + 1e-16f), t3 = logf(lh / ah + 1e-16f);
      float t4 = la * (PI_F / 180.f);
      int key = ((lvl * 3 + bn) * 128 + tj) * 128 + ti;
      int slot = -1;
      for (int e = 0; e < nent; e++) if (keys[e] == key) slot = e;
      if (slot < 0) { slot = nent++; keys[slot] = key; }
      lv_[slot] = lvl; an_[slot] = bn; tj_[slot] = tj; ti_[slot] = ti;
      tv[slot][0] = t0; tv[slot][1] = t1; tv[slot][2] = t2;
      tv[slot][3] = t3; tv[slot][4] = t4;
    }
    for (int e = 0; e < nent; e++) {
      int lvl = lv_[e], a = an_[e], tj = tj_[e], ti = ti_[e];
      const float* det = (lvl == 0) ? detL : ((lvl == 1) ? detM : detS);
      int ns = (lvl == 0) ? 19 : ((lvl == 1) ? 38 : 76);
      int s2 = ns * ns;
      int base = (b * 18 + a * 6) * s2 + tj * ns + ti;
      float p0 = det[base], p1 = det[base + s2];
      float p2 = det[base + 2 * s2], p3 = det[base + 3 * s2];
      float p4 = det[base + 4 * s2], p5 = det[base + 5 * s2];
      float lxy = bce_(p0, tv[e][0]) + bce_(p1, tv[e][1]);
      float d2 = p2 - tv[e][2], d3 = p3 - tv[e][3];
      float lwh = d2 * d2 + d3 * d3;
      float ang = sigmoidf_(p4) * (2.f * PI_F) - PI_F;
      float d = ang - tv[e][4];
      d = d - (2.f * PI_F) * rintf(d * (1.f / (2.f * PI_F)));
      float lang = fabsf(d);
      float pen = 1.f;
      if (lvl == 0) pen = (float)penL[(b * 3 + a) * 361 + tj * 19 + ti];
      float corr = bce_(p5, 1.f) - pen * softplus_(p5);
      total += lxy + 0.5f * lwh + lang + corr;
    }
  }
  float bs = block_reduce(total);
  if (threadIdx.x == 0) partials[OFF_POS] = bs;
}

// Final reduction: one full wave (EXEC all-1s). D = A(data 16x4) x B(ones 4x16) + C
// -> every column of D holds row-sums; exact f32 products, RNE accumulation.
typedef float    v2f  __attribute__((ext_vector_type(2)));
typedef float    v8f  __attribute__((ext_vector_type(8)));
typedef _Float16 v16h __attribute__((ext_vector_type(16)));

__global__ void k_final_reduce(const float* __restrict__ partials, float* out) {
  int lane = threadIdx.x;  // 0..31
  v8f acc = {};
#if __has_builtin(__builtin_amdgcn_wmma_f32_16x16x4_f32)
  v2f onesv; onesv[0] = 1.f; onesv[1] = 1.f;
  for (int base = 0; base < NPART; base += 64) {
    v2f a;
    a[0] = partials[base + lane];        // A[m][0] / A[m][2]
    a[1] = partials[base + 32 + lane];   // A[m][1] / A[m][3]
    acc = __builtin_amdgcn_wmma_f32_16x16x4_f32(false, a, false, onesv,
                                                (short)0, acc, false, false);
  }
#else
  v16h onesh;
  #pragma unroll
  for (int i = 0; i < 16; i++) onesh[i] = (_Float16)1.0f;
  for (int base = 0; base < NPART; base += 512) {
    v16h a;
    #pragma unroll
    for (int i = 0; i < 16; i++) a[i] = (_Float16)partials[base + i * 32 + lane];
    acc = __builtin_amdgcn_wmma_f32_16x16x32_f16(false, a, false, onesh,
                                                 (short)0, acc, false, false);
  }
#endif
  // lane<16 holds M=r, lane>=16 holds M=8+r in VGPR r; columns identical.
  float s = acc[0] + acc[1] + acc[2] + acc[3] + acc[4] + acc[5] + acc[6] + acc[7];
  s += __shfl_xor(s, 16, 32);
  if (lane == 0) out[0] = s;
}

// ---------------- launch ----------------
extern "C" void kernel_launch(void* const* d_in, const int* in_sizes, int n_in,
                              void* d_out, int out_size, void* d_ws, size_t ws_size,
                              hipStream_t stream) {
  const float* detL   = (const float*)d_in[0];  // (64,18,19,19)
  const float* detM   = (const float*)d_in[1];  // (64,18,38,38)
  const float* detS   = (const float*)d_in[2];  // (64,18,76,76)
  const float* labels = (const float*)d_in[3];  // (64,20,5)

  char* ws = (char*)d_ws;
  int* cnt            = (int*)(ws + WS_CNT);
  int* validany       = (int*)(ws + WS_VALIDANY);
  unsigned char* penL = (unsigned char*)(ws + WS_PENMASK);
  float* partials     = (float*)(ws + WS_PARTIALS);

  k_init<<<(NPART + 255) / 256, 256, 0, stream>>>(cnt, validany, partials);
  k_prep<<<1, 64, 0, stream>>>(labels, validany);
  k_countL<<<(64 * 1083 + 255) / 256, 256, 0, stream>>>(detL, cnt);
  k_negL<<<dim3(5, 64), 256, 0, stream>>>(detL, labels, cnt, validany, penL, partials);
  k_neg_conf<<<1083, 256, 0, stream>>>(detM, partials, OFF_M, 38);  // 64*3*38*38 cells
  k_neg_conf<<<4333, 256, 0, stream>>>(detS, partials, OFF_S, 76);  // 64*3*76*76 cells
  k_pos<<<1, 64, 0, stream>>>(detL, detM, detS, labels, penL, partials);
  k_final_reduce<<<1, 32, 0, stream>>>(partials, (float*)d_out);
}
